// GCN_5600637354092
// MI455X (gfx1250) — compile-verified, matrix operand order checked by reading, not run on hardware
//
#include <hip/hip_runtime.h>

// ---------------------------------------------------------------------------
// Types for WMMA (CDNA5 / gfx1250, wave32)
// ---------------------------------------------------------------------------
typedef __attribute__((ext_vector_type(16))) __bf16          v16bf;
typedef __attribute__((ext_vector_type(8)))  float           v8f;

// fp32 -> bf16 bits, round-to-nearest-even
__device__ __forceinline__ unsigned int f2bf(float f) {
    unsigned int u = __float_as_uint(f);
    unsigned int r = u + 0x7FFFu + ((u >> 16) & 1u);
    return r >> 16;
}

__device__ __forceinline__ uint2 pack4bf(float a, float b, float c, float d) {
    uint2 p;
    p.x = f2bf(a) | (f2bf(b) << 16);
    p.y = f2bf(c) | (f2bf(d) << 16);
    return p;
}

// ---------------------------------------------------------------------------
// Degree / norm kernels
// ---------------------------------------------------------------------------
__global__ void set_deg_one(float* __restrict__ deg, int n) {
    int i = blockIdx.x * blockDim.x + threadIdx.x;
    if (i < n) deg[i] = 1.0f;   // self-loop
}

__global__ void edge_deg(const int* __restrict__ dst, float* __restrict__ deg, int e) {
    int i = blockIdx.x * blockDim.x + threadIdx.x;
    if (i < e) atomicAdd(&deg[dst[i]], 1.0f);
}

__global__ void compute_dinv(const float* __restrict__ deg, float* __restrict__ dinv, int n) {
    int i = blockIdx.x * blockDim.x + threadIdx.x;
    if (i < n) {
        float d = deg[i];
        dinv[i] = (d > 0.0f) ? rsqrtf(d) : 0.0f;
    }
}

// ---------------------------------------------------------------------------
// Streaming fp32 -> bf16 convert (vectorized: float4 in, 8 bytes out)
// ---------------------------------------------------------------------------
__global__ void cvt_bf16(const float* __restrict__ in, unsigned short* __restrict__ out,
                         long long nvec4) {
    long long t = (long long)blockIdx.x * blockDim.x + threadIdx.x;
    if (t >= nvec4) return;
    float4 v = *(const float4*)(in + t * 4);
    *(uint2*)(out + t * 4) = pack4bf(v.x, v.y, v.z, v.w);
}

// ---------------------------------------------------------------------------
// Weight transpose + convert: W[K][NOUT] fp32 -> Wt[NOUT][K] bf16
// ---------------------------------------------------------------------------
__global__ void transpose_bf16(const float* __restrict__ W, unsigned short* __restrict__ Wt,
                               int K, int NOUT) {
    int idx = blockIdx.x * blockDim.x + threadIdx.x;
    if (idx >= K * NOUT) return;
    int k = idx / NOUT;
    int n = idx % NOUT;
    Wt[(size_t)n * K + k] = (unsigned short)f2bf(W[idx]);
}

// ---------------------------------------------------------------------------
// WMMA GEMM: Out[N][NOUT] = A[N][K] (bf16) x Wt[NOUT][K] (bf16), fp32 acc.
// Block = NOUT/16 waves, computes 64 rows x NOUT cols (4 row-subtiles with
// separate accumulators so each B fragment is loaded once per k-step and
// reused 4x -> 4x less L2 traffic on the B panel). A tile staged in LDS with
// a +8 bf16 row pad (row stride 528B -> consecutive rows 4 banks apart).
// A-matrix per-lane layout (16-bit, 16x32): lanes 0-15 row m=lane&15, elems
// j=0..7 -> K=base..base+7, j=8..15 -> K=base+16..base+23, base=(lane>=16)*8.
// B symmetric, column n = lane&15. C/D: vgpr r, lane l -> M=r+8*(l>>4), N=l&15.
// ---------------------------------------------------------------------------
template <int K, int NOUT>
__global__ void gemm_wmma_bf16(const unsigned short* __restrict__ A,
                               const unsigned short* __restrict__ Wt,
                               float* __restrict__ Out, int nrows) {
    constexpr int LDA = K + 8;                       // padded row (ushorts)
    constexpr int MR  = 64;                          // rows per block
    __shared__ __align__(16) unsigned short sA[MR * LDA];

    const int wave = threadIdx.x >> 5;
    const int lane = threadIdx.x & 31;
    const int m0   = blockIdx.x * MR;
    const int n0   = wave * 16;
    const int lo   = lane & 15;
    const int hi   = lane >> 4;
    const int base = hi * 8;

    // ---- cooperative load of the MR x K bf16 A tile into LDS ----
    constexpr int CH = (MR * K) / 8;                 // 8-element (16B) chunks
#pragma unroll
    for (int ch = threadIdx.x; ch < CH; ch += NOUT * 2) {
        int r   = ch / (K / 8);
        int cc  = (ch % (K / 8)) * 8;
        int row = m0 + r;
        if (row >= nrows) row = nrows - 1;           // clamp (partial last block)
        *(uint4*)&sA[r * LDA + cc] = *(const uint4*)(A + (size_t)row * K + cc);
    }
    __syncthreads();

    v8f acc[4] = {v8f{}, v8f{}, v8f{}, v8f{}};
    const unsigned short* brow = Wt + (size_t)(n0 + lo) * K + base;
    const int aoff = lo * LDA + base;

#pragma unroll
    for (int kb = 0; kb < K; kb += 32) {
        union { uint4 q[2]; v16bf b; } cb;
        cb.q[0] = *(const uint4*)(brow + kb);        // global_load_b128 (once)
        cb.q[1] = *(const uint4*)(brow + kb + 16);
#pragma unroll
        for (int t = 0; t < 4; ++t) {                // reuse B across 4 subtiles
            union { uint4 q[2]; v16bf b; } ca;
            ca.q[0] = *(const uint4*)(sA + t * 16 * LDA + aoff + kb);
            ca.q[1] = *(const uint4*)(sA + t * 16 * LDA + aoff + kb + 16);
            acc[t] = __builtin_amdgcn_wmma_f32_16x16x32_bf16(
                /*neg_a=*/false, ca.b, /*neg_b=*/false, cb.b,
                /*c_mod=*/(short)0, acc[t], /*reuse_a=*/false, /*reuse_b=*/false);
        }
    }

#pragma unroll
    for (int t = 0; t < 4; ++t) {
        int rbase = m0 + t * 16;
        if (rbase < nrows) {                         // block-uniform guard
            float* orow = Out + (size_t)(rbase + hi * 8) * NOUT + n0 + lo;
#pragma unroll
            for (int r = 0; r < 8; ++r)
                orow[(size_t)r * NOUT] = acc[t][r];
        }
    }
}

// ---------------------------------------------------------------------------
// Aggregation: self-loop term (also initializes agg), then edge scatter-add.
// ---------------------------------------------------------------------------
template <int F>
__global__ void agg_self(const float* __restrict__ h, const float* __restrict__ dinv,
                         float* __restrict__ agg, int n) {
    constexpr int C = F / 4;
    long long t = (long long)blockIdx.x * blockDim.x + threadIdx.x;
    if (t >= (long long)n * C) return;
    int i = (int)(t / C);
    int c = (int)(t % C);
    float di = dinv[i];
    float w  = di * di;
    const float4 v = *(const float4*)(h + (size_t)i * F + c * 4);
    float4 o;
    o.x = w * v.x; o.y = w * v.y; o.z = w * v.z; o.w = w * v.w;
    *(float4*)(agg + (size_t)i * F + c * 4) = o;
}

// One wave per edge; lane covers F/32 consecutive features.
template <int F>
__global__ void agg_edges(const int* __restrict__ src, const int* __restrict__ dst,
                          const float* __restrict__ dinv, const float* __restrict__ h,
                          float* __restrict__ agg, int E) {
    long long gid = (long long)blockIdx.x * blockDim.x + threadIdx.x;
    int e    = (int)(gid >> 5);
    int lane = (int)(gid & 31);
    if (e >= E) return;
    int s = src[e];
    int d = dst[e];
    float w = dinv[s] * dinv[d];
    constexpr int PER = F / 32;
    const float* hp = h   + (size_t)s * F + lane * PER;
    float*       ap = agg + (size_t)d * F + lane * PER;
    if constexpr (PER == 4) {
        float4 v = *(const float4*)hp;
        atomicAdd(ap + 0, w * v.x);
        atomicAdd(ap + 1, w * v.y);
        atomicAdd(ap + 2, w * v.z);
        atomicAdd(ap + 3, w * v.w);
    } else {
        float2 v = *(const float2*)hp;
        atomicAdd(ap + 0, w * v.x);
        atomicAdd(ap + 1, w * v.y);
    }
}

// ---------------------------------------------------------------------------
// Post-ops
// ---------------------------------------------------------------------------
// relu(agg + b) -> bf16 (feeds GEMM2 directly); thread handles 4 features.
__global__ void bias_relu_bf16(const float* __restrict__ agg, const float* __restrict__ b,
                               unsigned short* __restrict__ out, int n, int F) {
    long long t = (long long)blockIdx.x * blockDim.x + threadIdx.x;
    int C = F / 4;
    if (t >= (long long)n * C) return;
    int i = (int)(t / C);
    int c = (int)(t % C) * 4;
    float4 v  = *(const float4*)(agg + (size_t)i * F + c);
    float4 bb = *(const float4*)(b + c);
    float r0 = fmaxf(v.x + bb.x, 0.0f);
    float r1 = fmaxf(v.y + bb.y, 0.0f);
    float r2 = fmaxf(v.z + bb.z, 0.0f);
    float r3 = fmaxf(v.w + bb.w, 0.0f);
    *(uint2*)(out + (size_t)i * F + c) = pack4bf(r0, r1, r2, r3);
}

// Wave-per-row log_softmax over 64 features (lane handles f=lane, f=lane+32).
__global__ void logsoftmax64(const float* __restrict__ agg, const float* __restrict__ b,
                             float* __restrict__ out, int n) {
    long long gid = (long long)blockIdx.x * blockDim.x + threadIdx.x;
    int row  = (int)(gid >> 5);
    int lane = (int)(gid & 31);
    if (row >= n) return;
    float x0 = agg[(size_t)row * 64 + lane]      + b[lane];
    float x1 = agg[(size_t)row * 64 + lane + 32] + b[lane + 32];
    float m = fmaxf(x0, x1);
#pragma unroll
    for (int o = 16; o > 0; o >>= 1) m = fmaxf(m, __shfl_xor(m, o, 32));
    float s = expf(x0 - m) + expf(x1 - m);
#pragma unroll
    for (int o = 16; o > 0; o >>= 1) s += __shfl_xor(s, o, 32);
    float ls = logf(s);
    out[(size_t)row * 64 + lane]      = x0 - m - ls;
    out[(size_t)row * 64 + lane + 32] = x1 - m - ls;
}

// ---------------------------------------------------------------------------
// Launch
// ---------------------------------------------------------------------------
extern "C" void kernel_launch(void* const* d_in, const int* in_sizes, int n_in,
                              void* d_out, int out_size, void* d_ws, size_t ws_size,
                              hipStream_t stream) {
    const float* x   = (const float*)d_in[0];   // [N,256]
    const float* W1  = (const float*)d_in[1];   // [256,128]
    const float* b1  = (const float*)d_in[2];   // [128]
    const float* W2  = (const float*)d_in[3];   // [128,64]
    const float* b2  = (const float*)d_in[4];   // [64]
    const int*   ei  = (const int*)d_in[5];     // [2,E]

    const int N = in_sizes[0] / 256;            // 100000 (divisible by 16)
    const int E = in_sizes[5] / 2;              // 3200000
    const int* src = ei;
    const int* dst = ei + E;

    // -------- workspace carve-out (buffers reused across phases) --------
    char* ws = (char*)d_ws;
    size_t o = 0;
    auto carve = [&](size_t bytes) -> size_t {
        size_t r = o;
        o += (bytes + 255) & ~(size_t)255;
        return r;
    };
    float*          deg   = (float*)(ws + carve((size_t)N * 4));
    float*          dinv  = (float*)(ws + carve((size_t)N * 4));
    unsigned short* W1t   = (unsigned short*)(ws + carve((size_t)128 * 256 * 2));
    unsigned short* W2t   = (unsigned short*)(ws + carve((size_t)64 * 128 * 2));
    // 51.2MB: X-bf16 [N][256], later reused as H1act-bf16 [N][128]
    unsigned short* bfBuf = (unsigned short*)(ws + carve((size_t)N * 256 * 2));
    // 51.2MB: H1lin fp32 [N][128], later H2lin fp32 [N][64]
    float*          bufA  = (float*)(ws + carve((size_t)N * 128 * 4));
    // 51.2MB: H1agg fp32 [N][128], later H2agg fp32 [N][64]
    float*          bufB  = (float*)(ws + carve((size_t)N * 128 * 4));

    const int B = 256;
    const int gemmGrid = (N + 63) / 64;

    // -------- gcn_norm: deg, dinv --------
    set_deg_one<<<(N + B - 1) / B, B, 0, stream>>>(deg, N);
    edge_deg<<<(E + B - 1) / B, B, 0, stream>>>(dst, deg, E);
    compute_dinv<<<(N + B - 1) / B, B, 0, stream>>>(deg, dinv, N);

    // -------- weights: transpose + bf16 (one-time, tiny) --------
    transpose_bf16<<<(256 * 128 + B - 1) / B, B, 0, stream>>>(W1, W1t, 256, 128);
    transpose_bf16<<<(128 * 64 + B - 1) / B, B, 0, stream>>>(W2, W2t, 128, 64);

    // -------- x -> bf16 (streaming, one pass) --------
    {
        long long nv4 = (long long)N * 256 / 4;
        cvt_bf16<<<(int)((nv4 + B - 1) / B), B, 0, stream>>>(x, bfBuf, nv4);
    }

    // -------- layer 1: H1lin = x @ W1 (WMMA bf16, fp32 acc) --------
    gemm_wmma_bf16<256, 128><<<gemmGrid, 256, 0, stream>>>(bfBuf, W1t, bufA, N);
    // aggregation: agg = D^-1/2 A_hat D^-1/2 * H1lin
    agg_self<128><<<(int)(((long long)N * 32 + B - 1) / B), B, 0, stream>>>(bufA, dinv, bufB, N);
    agg_edges<128><<<(int)(((long long)E * 32 + B - 1) / B), B, 0, stream>>>(src, dst, dinv, bufA, bufB, E);
    // H1act = relu(agg + b1) -> bf16 (reuses the X-bf16 region, now dead)
    bias_relu_bf16<<<(int)(((long long)N * 32 + B - 1) / B), B, 0, stream>>>(bufB, b1, bfBuf, N, 128);

    // -------- layer 2: H2lin = H1act @ W2 --------
    gemm_wmma_bf16<128, 64><<<gemmGrid, 128, 0, stream>>>(bfBuf, W2t, bufA, N);
    agg_self<64><<<(int)(((long long)N * 16 + B - 1) / B), B, 0, stream>>>(bufA, dinv, bufB, N);
    agg_edges<64><<<(int)(((long long)E * 32 + B - 1) / B), B, 0, stream>>>(src, dst, dinv, bufA, bufB, E);

    // -------- log_softmax --------
    logsoftmax64<<<(int)(((long long)N * 32 + B - 1) / B), B, 0, stream>>>(bufB, b2, (float*)d_out, N);
}
